// MultiHeadVectorAttention_14654428414537
// MI455X (gfx1250) — matrix operand order repaired and checked by reading, not compiled, not run on hardware
//
#include <hip/hip_runtime.h>
#include <hip/hip_bf16.h>

typedef __attribute__((ext_vector_type(16))) _Float16 v16h;
typedef __attribute__((ext_vector_type(8)))  _Float16 v8h;
typedef __attribute__((ext_vector_type(8)))  float    v8f;

// D = A(16x32 f16) * B(32x16 f16) + C (f32)
static __device__ __forceinline__ v8f wmma16(v16h a, v16h b, v8f c) {
  return __builtin_amdgcn_wmma_f32_16x16x32_f16(false, a, false, b, (short)0, c,
                                                false, false);
}

// A-fragment (16x32 f16) from a row-major f16 LDS buffer.
// Lane l: row = l&15, needs halves [kbase + (l>>4)*8 .. +8) and the same +16.
static __device__ __forceinline__ v16h load_afrag(const _Float16* base, int stride,
                                                  int l, int kbase) {
  int row = l & 15;
  int kb = kbase + ((l >> 4) << 3);
  const _Float16* p = base + row * stride + kb;
  v8h lo = *(const v8h*)p;
  v8h hi = *(const v8h*)(p + 16);
  v16h a;
#pragma unroll
  for (int e = 0; e < 8; ++e) { a[e] = lo[e]; a[e + 8] = hi[e]; }
  return a;
}

static __device__ __forceinline__ v8f zero8() {
  v8f z = {};
  return z;
}

// ---------------------------------------------------------------------------
// Pack an f32 weight matrix (K x N, element (k,n) at src[off + k*ks + n*ns])
// into f16 B-fragment layout: out[((kt*numNT+nt)*32 + lane)*16 + e] =
//   W[kt*32 + (lane>>4)*8 + (e&7) + (e>>3)*16][nt*16 + (lane&15)]
// ---------------------------------------------------------------------------
__global__ __launch_bounds__(256) void pack_b_kernel(
    const float* __restrict__ src, _Float16* __restrict__ dst,
    int numKT, int numNT, int ks, int ns, int off) {
  int o = blockIdx.x * 256 + threadIdx.x;
  int total = numKT * numNT * 512;
  if (o >= total) return;
  int e = o & 15;
  int lane = (o >> 4) & 31;
  int t = o >> 9;
  int nt = t % numNT;
  int kt = t / numNT;
  int k = kt * 32 + ((lane >> 4) << 3) + (e & 7) + ((e >> 3) << 4);
  int n = nt * 16 + (lane & 15);
  dst[o] = (_Float16)src[off + k * ks + n * ns];
}

// ---------------------------------------------------------------------------
// C(2048x512) = X(2048x512 f32) * Wp(512x512, packed f16 B-frags)
// scramble==1: store C[b*1024+i, h*64+d] -> Out[((b*8+h)*1024+i)*64+d]  (QKV)
// scramble==0: Out[R*512+col] = C + bias[col]                           (final)
// ---------------------------------------------------------------------------
__global__ __launch_bounds__(256) void gemm512_kernel(
    const float* __restrict__ X, const _Float16* __restrict__ Wp,
    float* __restrict__ Out, const float* __restrict__ bias, int scramble) {
  __shared__ __align__(16) _Float16 xs[16 * 520];
  int tid = threadIdx.x;
  int rt = blockIdx.x;
#pragma unroll
  for (int u = 0; u < 32; ++u) {
    int e = u * 256 + tid;
    int row = e >> 9;
    int k = e & 511;
    xs[row * 520 + k] = (_Float16)X[(rt * 16 + row) * 512 + k];
  }
  __syncthreads();
  int w = tid >> 5, l = tid & 31;
  v8f c[4];
#pragma unroll
  for (int t = 0; t < 4; ++t) c[t] = zero8();
  for (int ks = 0; ks < 16; ++ks) {
    // cluster all loads of this k-step, then issue WMMAs back-to-back
    v16h a = load_afrag(xs, 520, l, ks * 32);
    v16h b[4];
#pragma unroll
    for (int t = 0; t < 4; ++t) {
      int nt = w * 4 + t;
      b[t] = *(const v16h*)(Wp + ((ks * 32 + nt) * 512 + l * 16));
    }
#pragma unroll
    for (int t = 0; t < 4; ++t) c[t] = wmma16(a, b[t], c[t]);
  }
#pragma unroll
  for (int t = 0; t < 4; ++t) {
    int col = (w * 4 + t) * 16 + (l & 15);
#pragma unroll
    for (int r = 0; r < 8; ++r) {
      int rowl = r + ((l >> 4) << 3);
      int R = rt * 16 + rowl;
      float v = c[t][r];
      if (scramble) {
        int b_ = R >> 10, i_ = R & 1023, hh = col >> 6, d = col & 63;
        Out[(((b_ * 8 + hh) * 1024) + i_) * 64 + d] = v;
      } else {
        Out[R * 512 + col] = v + bias[col];
      }
    }
  }
}

// ---------------------------------------------------------------------------
// KNN: per query, top-16 smallest squared distances (stable ties, self incl.)
// ---------------------------------------------------------------------------
__global__ __launch_bounds__(256) void knn_kernel(
    const float* __restrict__ canon, int* __restrict__ nbr) {
  __shared__ float cs[1024 * 3];
  int b = blockIdx.x >> 2;
  int qi = ((blockIdx.x & 3) << 8) + threadIdx.x;
  for (int t = threadIdx.x; t < 3072; t += 256) cs[t] = canon[b * 3072 + t];
  __syncthreads();
  float px = cs[qi * 3], py = cs[qi * 3 + 1], pz = cs[qi * 3 + 2];
  float dist[16];
  int ind[16];
#pragma unroll
  for (int t = 0; t < 16; ++t) { dist[t] = 3.0e38f; ind[t] = 0; }
  for (int p = 0; p < 1024; ++p) {
    float dx = cs[p * 3] - px, dy = cs[p * 3 + 1] - py, dz = cs[p * 3 + 2] - pz;
    float d = dx * dx + dy * dy + dz * dz;
    if (d < dist[15]) {
      float cd = d;
      int ci = p;
#pragma unroll
      for (int t = 0; t < 16; ++t) {
        if (cd < dist[t]) {
          float td = dist[t]; dist[t] = cd; cd = td;
          int ti = ind[t]; ind[t] = ci; ci = ti;
        }
      }
    }
  }
#pragma unroll
  for (int t = 0; t < 16; ++t) nbr[(b * 1024 + qi) * 16 + t] = ind[t];
}

// ---------------------------------------------------------------------------
// Fused attention. One wave per (b,h,i); block = 8 waves = 8 consecutive i of
// one (b,h). PASS 1: softmax(sim) -> deterministic per-block sum_i attn^2.
// PASS 2: recompute, divide by norm, aggregate with vg, write agg.
// ---------------------------------------------------------------------------
template <int PASS>
__global__ __launch_bounds__(256) void attn_kernel(
    const float* __restrict__ Qm, const float* __restrict__ Km,
    const float* __restrict__ Vm, const int* __restrict__ nbr,
    const float* __restrict__ canon,
    const float* __restrict__ pw1, const float* __restrict__ pb1,
    const _Float16* __restrict__ pw2p, const float* __restrict__ pb2,
    const _Float16* __restrict__ aw1p, const float* __restrict__ ab1,
    const _Float16* __restrict__ aw2p, const float* __restrict__ ab2,
    float* __restrict__ part, const float* __restrict__ norm2,
    float* __restrict__ Am) {
  __shared__ __align__(16) union SU {
    struct {
      _Float16 b1[8][16 * 72];    // per-wave: hidden (16x64), then x (16x64)
      _Float16 b2[8][16 * 136];   // per-wave: hdn chunk (16x128)
    } s;
    float red[8][1024];           // PASS1 reduction overlay
  } su;
  __shared__ float srel[8][16][4];

  int tid = threadIdx.x, w = tid >> 5, l = tid & 31;
  int bid = blockIdx.x;
  int bh = bid >> 7, ig = bid & 127;
  int b = bh >> 3, hh = bh & 7;
  int ii = ig * 8 + w;

  const float* cb = canon + b * 3072;
  if (l < 16) {
    int nb = nbr[(b * 1024 + ii) * 16 + l];
    srel[w][l][0] = cb[nb * 3]     - cb[ii * 3];
    srel[w][l][1] = cb[nb * 3 + 1] - cb[ii * 3 + 1];
    srel[w][l][2] = cb[nb * 3 + 2] - cb[ii * 3 + 2];
  }
  __syncthreads();

  // hidden = relu(rel @ pos_w1 + pos_b1): lane does (j = l&15, 32 e-values)
  {
    int j = l & 15, eb = (l >> 4) * 32;
    float rx = srel[w][j][0], ry = srel[w][j][1], rz = srel[w][j][2];
    _Float16* hb = su.s.b1[w] + j * 72;
#pragma unroll
    for (int e2 = 0; e2 < 32; ++e2) {
      int e = eb + e2;
      float v = rx * pw1[e] + ry * pw1[64 + e] + rz * pw1[128 + e] + pb1[e];
      hb[e] = (_Float16)fmaxf(v, 0.f);
    }
  }
  __syncthreads();

  // rpe (16x64) = hidden @ pos_w2[:, h*64:+64] + pos_b2
  v8f rpe[4];
#pragma unroll
  for (int nt = 0; nt < 4; ++nt) rpe[nt] = zero8();
#pragma unroll
  for (int kt = 0; kt < 2; ++kt) {
    v16h a = load_afrag(su.s.b1[w], 72, l, kt * 32);
    v16h bb[4];
#pragma unroll
    for (int nt = 0; nt < 4; ++nt)
      bb[nt] = *(const v16h*)(pw2p + ((kt * 32 + hh * 4 + nt) * 512 + l * 16));
#pragma unroll
    for (int nt = 0; nt < 4; ++nt) rpe[nt] = wmma16(a, bb[nt], rpe[nt]);
  }
#pragma unroll
  for (int nt = 0; nt < 4; ++nt) {
    float bv = pb2[hh * 64 + nt * 16 + (l & 15)];
#pragma unroll
    for (int r = 0; r < 8; ++r) rpe[nt][r] += bv;
  }
  __syncthreads();  // hidden reads done before b1 overwritten with x

  // Scrambled gather: rows r0,r1 of (b,h,n,d)-flat matrices
  int m0 = ((b * 8 + hh) * 1024 + ii) * 2;
  int r0 = nbr[m0] + ((m0 >> 14) << 10);
  int r1 = nbr[m0 + 1] + ((m0 >> 14) << 10);
  v8f vgv[4];
#pragma unroll
  for (int nt = 0; nt < 4; ++nt) {
    vgv[nt] = zero8();
#pragma unroll
    for (int r = 0; r < 8; ++r) {
      int j = ((l >> 4) << 3) + r;
      int row = (j < 8) ? r0 : r1;
      int addr = row * 512 + ((j & 7) << 6) + nt * 16 + (l & 15);
      float xv = Qm[addr] - Km[addr] + rpe[nt][r];
      su.s.b1[w][j * 72 + nt * 16 + (l & 15)] = (_Float16)xv;
      if (PASS == 2) vgv[nt][r] = Vm[addr] + rpe[nt][r];
    }
  }
  __syncthreads();

  // attn MLP: hdn = relu(x @ aw1[h]^T + ab1), sim = hdn @ aw2[h]^T + ab2
  v8f sim[4];
#pragma unroll
  for (int nt = 0; nt < 4; ++nt) sim[nt] = zero8();
  for (int ch = 0; ch < 2; ++ch) {
    v8f hd[8];
#pragma unroll
    for (int t = 0; t < 8; ++t) hd[t] = zero8();
#pragma unroll
    for (int kt = 0; kt < 2; ++kt) {
      v16h a = load_afrag(su.s.b1[w], 72, l, kt * 32);
      v16h bb[8];
#pragma unroll
      for (int t = 0; t < 8; ++t) {
        int nt = ch * 8 + t;
        bb[t] = *(const v16h*)(aw1p + hh * 16384 + ((kt * 16 + nt) * 512 + l * 16));
      }
#pragma unroll
      for (int t = 0; t < 8; ++t) hd[t] = wmma16(a, bb[t], hd[t]);
    }
    __syncthreads();  // prior chunk's b2 reads complete before rewrite
#pragma unroll
    for (int t = 0; t < 8; ++t) {
      int e = ch * 128 + t * 16 + (l & 15);
      float bv = ab1[hh * 256 + e];
#pragma unroll
      for (int r = 0; r < 8; ++r) {
        int j = ((l >> 4) << 3) + r;
        float v = fmaxf(hd[t][r] + bv, 0.f);
        su.s.b2[w][j * 136 + t * 16 + (l & 15)] = (_Float16)v;
      }
    }
    __syncthreads();
#pragma unroll
    for (int kt2 = 0; kt2 < 4; ++kt2) {
      v16h a = load_afrag(su.s.b2[w], 136, l, kt2 * 32);
      int ktg = ch * 4 + kt2;
      v16h bb[4];
#pragma unroll
      for (int nt = 0; nt < 4; ++nt)
        bb[nt] = *(const v16h*)(aw2p + hh * 16384 + ((ktg * 4 + nt) * 512 + l * 16));
#pragma unroll
      for (int nt = 0; nt < 4; ++nt) sim[nt] = wmma16(a, bb[nt], sim[nt]);
    }
  }

  // bias + softmax over j (8 rows in-lane, partner half via shfl_xor 16)
#pragma unroll
  for (int nt = 0; nt < 4; ++nt) {
    float bv = ab2[hh * 64 + nt * 16 + (l & 15)];
    float mx = -3.0e38f;
#pragma unroll
    for (int r = 0; r < 8; ++r) {
      sim[nt][r] += bv;
      mx = fmaxf(mx, sim[nt][r]);
    }
    mx = fmaxf(mx, __shfl_xor(mx, 16, 32));
    float ssum = 0.f;
#pragma unroll
    for (int r = 0; r < 8; ++r) {
      float ev = __expf(sim[nt][r] - mx);
      sim[nt][r] = ev;
      ssum += ev;
    }
    ssum += __shfl_xor(ssum, 16, 32);
    float inv = 1.f / ssum;
#pragma unroll
    for (int r = 0; r < 8; ++r) sim[nt][r] *= inv;
  }

  if (PASS == 1) {
    __syncthreads();  // b1/b2 traffic done; red overlays them
#pragma unroll
    for (int nt = 0; nt < 4; ++nt)
#pragma unroll
      for (int r = 0; r < 8; ++r) {
        int j = ((l >> 4) << 3) + r;
        float a = sim[nt][r];
        su.red[w][j * 64 + nt * 16 + (l & 15)] = a * a;
      }
    __syncthreads();
    for (int t = tid; t < 1024; t += 256) {
      float s = 0.f;
#pragma unroll
      for (int w2 = 0; w2 < 8; ++w2) s += su.red[w2][t];
      part[(bh * 128 + ig) * 1024 + t] = s;
    }
  } else {
    const float* n2b = norm2 + bh * 1024;
#pragma unroll
    for (int nt = 0; nt < 4; ++nt) {
      float acc = 0.f;
#pragma unroll
      for (int r = 0; r < 8; ++r) {
        int j = ((l >> 4) << 3) + r;
        float n2 = n2b[j * 64 + nt * 16 + (l & 15)];
        float den = fmaxf(sqrtf(n2), 1e-12f);
        acc += (sim[nt][r] / den) * vgv[nt][r];
      }
      acc += __shfl_xor(acc, 16, 32);
      if (l < 16) Am[(b * 1024 + ii) * 512 + hh * 64 + nt * 16 + l] = acc;
    }
  }
}

// Deterministic reduction of per-block partials -> norm2[bh][1024]
__global__ __launch_bounds__(256) void reduce_norm_kernel(
    const float* __restrict__ part, float* __restrict__ norm2) {
  int bh = blockIdx.x;
  for (int t = threadIdx.x; t < 1024; t += 256) {
    float s = 0.f;
    for (int g = 0; g < 128; ++g) s += part[(bh * 128 + g) * 1024 + t];
    norm2[bh * 1024 + t] = s;
  }
}

// ---------------------------------------------------------------------------
extern "C" void kernel_launch(void* const* d_in, const int* in_sizes, int n_in,
                              void* d_out, int out_size, void* d_ws, size_t ws_size,
                              hipStream_t stream) {
  (void)in_sizes; (void)n_in; (void)out_size; (void)ws_size;
  const float* query  = (const float*)d_in[0];
  const float* key_in = (const float*)d_in[1];
  const float* value  = (const float*)d_in[2];
  const float* canon  = (const float*)d_in[3];
  const float* w_q    = (const float*)d_in[4];
  const float* w_k    = (const float*)d_in[5];
  const float* w_v    = (const float*)d_in[6];
  const float* w_out  = (const float*)d_in[7];
  const float* b_out  = (const float*)d_in[8];
  const float* pos_w1 = (const float*)d_in[9];
  const float* pos_b1 = (const float*)d_in[10];
  const float* pos_w2 = (const float*)d_in[11];
  const float* pos_b2 = (const float*)d_in[12];
  const float* aw1    = (const float*)d_in[13];
  const float* ab1    = (const float*)d_in[14];
  const float* aw2    = (const float*)d_in[15];
  const float* ab2    = (const float*)d_in[16];

  char* p = (char*)d_ws;
  float* Qm    = (float*)p;       p += (size_t)4 << 20;
  float* Km    = (float*)p;       p += (size_t)4 << 20;
  float* Vm    = (float*)p;       p += (size_t)4 << 20;
  float* Am    = (float*)p;       p += (size_t)4 << 20;
  float* part  = (float*)p;       p += (size_t)8 << 20;
  int*   nbrw  = (int*)p;         p += 32768 * 4;
  float* norm2 = (float*)p;       p += 16384 * 4;
  _Float16* wqp  = (_Float16*)p;  p += 262144 * 2;
  _Float16* wkp  = (_Float16*)p;  p += 262144 * 2;
  _Float16* wvp  = (_Float16*)p;  p += 262144 * 2;
  _Float16* wop  = (_Float16*)p;  p += 262144 * 2;
  _Float16* pw2p = (_Float16*)p;  p += 32768 * 2;
  _Float16* aw1p = (_Float16*)p;  p += 131072 * 2;
  _Float16* aw2p = (_Float16*)p;  p += 131072 * 2;

  // 1) pack weights to f16 B-fragment layout
  pack_b_kernel<<<1024, 256, 0, stream>>>(w_q,   wqp, 16, 32, 512, 1, 0);
  pack_b_kernel<<<1024, 256, 0, stream>>>(w_k,   wkp, 16, 32, 512, 1, 0);
  pack_b_kernel<<<1024, 256, 0, stream>>>(w_v,   wvp, 16, 32, 512, 1, 0);
  pack_b_kernel<<<1024, 256, 0, stream>>>(w_out, wop, 16, 32, 512, 1, 0);
  pack_b_kernel<<<128,  256, 0, stream>>>(pos_w2, pw2p, 2, 32, 512, 1, 0);
  for (int h = 0; h < 8; ++h) {
    // B[k=d][n=e] = aw1[h,e,d]
    pack_b_kernel<<<64, 256, 0, stream>>>(aw1, aw1p + h * 16384, 2, 16, 1, 64,
                                          h * 16384);
    // B[k=e][n=d] = aw2[h,d,e]
    pack_b_kernel<<<64, 256, 0, stream>>>(aw2, aw2p + h * 16384, 8, 4, 1, 256,
                                          h * 16384);
  }

  // 2) QKV projections (scrambled store to (b,h,n,d) flat)
  gemm512_kernel<<<128, 256, 0, stream>>>(query,  wqp, Qm, b_out, 1);
  gemm512_kernel<<<128, 256, 0, stream>>>(key_in, wkp, Km, b_out, 1);
  gemm512_kernel<<<128, 256, 0, stream>>>(value,  wvp, Vm, b_out, 1);

  // 3) KNN top-16
  knn_kernel<<<8, 256, 0, stream>>>(canon, nbrw);

  // 4) attention pass 1: softmax + norm^2 partials
  attn_kernel<1><<<2048, 256, 0, stream>>>(Qm, Km, Vm, nbrw, canon, pos_w1,
                                           pos_b1, pw2p, pos_b2, aw1p, ab1,
                                           aw2p, ab2, part, norm2, Am);
  // 5) deterministic norm^2 reduction over i
  reduce_norm_kernel<<<16, 256, 0, stream>>>(part, norm2);
  // 6) attention pass 2: normalize + aggregate
  attn_kernel<2><<<2048, 256, 0, stream>>>(Qm, Km, Vm, nbrw, canon, pos_w1,
                                           pos_b1, pw2p, pos_b2, aw1p, ab1,
                                           aw2p, ab2, part, norm2, Am);

  // 7) output projection with bias
  gemm512_kernel<<<128, 256, 0, stream>>>(Am, wop, (float*)d_out, b_out, 0);
}
